// W4A16LinearStatic_47356309405799
// MI455X (gfx1250) — compile-verified, matrix operand order checked by reading, not run on hardware
//
#include <hip/hip_runtime.h>

// ---- vector types -----------------------------------------------------------
typedef __attribute__((ext_vector_type(16))) __bf16          v16bf;
typedef __attribute__((ext_vector_type(8)))  __bf16          bf16x8;
typedef __attribute__((ext_vector_type(8)))  float           v8f;
typedef __attribute__((ext_vector_type(8)))  unsigned short  us8;
typedef __attribute__((ext_vector_type(4)))  float           f4;
typedef __attribute__((ext_vector_type(4)))  int             i4;
typedef __attribute__((ext_vector_type(4)))  unsigned int    u32x4;
typedef __attribute__((ext_vector_type(8)))  int             i32x8;
typedef __attribute__((ext_vector_type(4)))  int             i32x4;

#define TM  128     // M tile per block
#define TN  128     // N tile per block
#define TK  32      // K step (bf16 WMMA K)
#define LDA 40      // padded LDS row stride (elements); 80B rows, 16B aligned

__device__ __forceinline__ unsigned short f2bf(float f) {
  // fp32 -> bf16, round-to-nearest-even
  unsigned int u = __float_as_uint(f);
  u += 0x7FFFu + ((u >> 16) & 1u);
  return (unsigned short)(u >> 16);
}

// =============================================================================
// Tensor Data Mover: 2D tile load, bf16 elements, padded LDS rows.
// D# layout per CDNA5 ISA.
// tile_d0 elems per row (x2B = multiple of 4B), tile_d1 rows.
// pad: after 16 DWORDs (=32 bf16 = one row) insert 4 DWORDs -> LDS stride 40.
// =============================================================================
__device__ __forceinline__ void tdm_load_tile_2d(const void* gsrc,
                                                 unsigned lds_off,
                                                 unsigned tensor_d0,
                                                 unsigned tensor_d1,
                                                 unsigned tile_d0,
                                                 unsigned tile_d1,
                                                 unsigned stride0) {
  const unsigned long long ga = (unsigned long long)(uintptr_t)gsrc;
  u32x4 g0;
  g0[0] = 1u;                                        // count=1, user descriptor
  g0[1] = lds_off;                                   // LDS byte address
  g0[2] = (unsigned)(ga & 0xFFFFFFFFu);              // global_addr[31:0]
  g0[3] = (unsigned)((ga >> 32) & 0x1FFFFFFu)        // global_addr[56:32]
        | (2u << 30);                                // type=2 ("image")
  i32x8 g1;
  g1[0] = (int)((1u << 16)                           // data_size = 2 bytes
              | (1u << 20)                           // pad_enable
              | (3u << 22)                           // pad_interval: 16 DWORDs
              | (3u << 25));                         // pad_amount:   4 DWORDs
  g1[1] = (int)((tensor_d0 & 0xFFFFu) << 16);        // tensor_dim0[15:0]
  g1[2] = (int)(((tensor_d0 >> 16) & 0xFFFFu)        // tensor_dim0[31:16]
              | ((tensor_d1 & 0xFFFFu) << 16));      // tensor_dim1[15:0]
  g1[3] = (int)(((tensor_d1 >> 16) & 0xFFFFu)        // tensor_dim1[31:16]
              | ((tile_d0 & 0xFFFFu) << 16));        // tile_dim0
  g1[4] = (int)(tile_d1 & 0xFFFFu);                  // tile_dim1 (tile_dim2=0)
  g1[5] = (int)stride0;                              // tensor_dim0_stride[31:0]
  g1[6] = 0;                                         // stride0 hi / stride1 lo
  g1[7] = 0;
  const i32x4 z4 = {0, 0, 0, 0};
#if defined(__clang_major__) && (__clang_major__ >= 23)
  const i32x8 z8 = {0, 0, 0, 0, 0, 0, 0, 0};
  __builtin_amdgcn_tensor_load_to_lds(g0, g1, z4, z4, z8, 0);
#else
  __builtin_amdgcn_tensor_load_to_lds(g0, g1, z4, z4, 0);
#endif
}

// =============================================================================
// Prep kernels: one-shot precision conversion (bandwidth bound)
// =============================================================================
__global__ __launch_bounds__(256)
void cvt_x_bf16(const float* __restrict__ x, unsigned short* __restrict__ xb,
                long long total) {
  const long long i = ((long long)blockIdx.x * 256 + threadIdx.x) * 8;
  if (i + 8 > total) return;
  f4 a = *(const f4*)(x + i);
  f4 b = *(const f4*)(x + i + 4);
  us8 o;
  o[0] = f2bf(a[0]); o[1] = f2bf(a[1]); o[2] = f2bf(a[2]); o[3] = f2bf(a[3]);
  o[4] = f2bf(b[0]); o[5] = f2bf(b[1]); o[6] = f2bf(b[2]); o[7] = f2bf(b[3]);
  *(us8*)(xb + i) = o;
}

__global__ __launch_bounds__(256)
void dequant_w_bf16(const int* __restrict__ w, const float* __restrict__ scale,
                    const float* __restrict__ offs, unsigned short* __restrict__ wb,
                    int K) {
  const int n = blockIdx.y;
  const int k = (blockIdx.x * 256 + threadIdx.x) * 8;
  if (k + 8 > K) return;
  const float sc = scale[n];
  const float of = offs[n];
  const long long base = (long long)n * K + k;
  i4 a = *(const i4*)(w + base);
  i4 b = *(const i4*)(w + base + 4);
  us8 o;
  o[0] = f2bf(((float)a[0] + of) * sc); o[1] = f2bf(((float)a[1] + of) * sc);
  o[2] = f2bf(((float)a[2] + of) * sc); o[3] = f2bf(((float)a[3] + of) * sc);
  o[4] = f2bf(((float)b[0] + of) * sc); o[5] = f2bf(((float)b[1] + of) * sc);
  o[6] = f2bf(((float)b[2] + of) * sc); o[7] = f2bf(((float)b[3] + of) * sc);
  *(us8*)(wb + base) = o;
}

// =============================================================================
// Main GEMM: bf16 tiles streamed global->LDS by the Tensor Data Mover,
// double-buffered, 8 v_wmma per wave per K step. Zero per-element VALU.
// =============================================================================
__global__ __launch_bounds__(256)
void gemm_bf16_tdm(const unsigned short* __restrict__ xb,
                   const unsigned short* __restrict__ wb,
                   const float* __restrict__ bias,
                   float* __restrict__ y,
                   int M, int N, int K)
{
  __shared__ __align__(16) unsigned short ldsA[2][TM * LDA];
  __shared__ __align__(16) unsigned short ldsB[2][TN * LDA];

  const int tid  = threadIdx.x;
  const int wave = tid >> 5;
  const int lane = tid & 31;

  const int mTile = blockIdx.y * TM;
  const int nTile = blockIdx.x * TN;

  const int mOff = (wave & 3) * 32;   // wave grid 4(M) x 2(N): 32x64 per wave
  const int nOff = (wave >> 2) * 64;

  const int g  = lane >> 4;           // WMMA half-wave selector
  const int mr = lane & 15;

  v8f acc[2][4] = {};

  const unsigned offA0 = (unsigned)(uintptr_t)&ldsA[0][0];
  const unsigned offA1 = (unsigned)(uintptr_t)&ldsA[1][0];
  const unsigned offB0 = (unsigned)(uintptr_t)&ldsB[0][0];
  const unsigned offB1 = (unsigned)(uintptr_t)&ldsB[1][0];

  auto issue_tiles = [&](int buf, int k0) {
    tdm_load_tile_2d(xb + (size_t)mTile * K + k0, buf ? offA1 : offA0,
                     (unsigned)K, (unsigned)M, TK, TM, (unsigned)K);
    tdm_load_tile_2d(wb + (size_t)nTile * K + k0, buf ? offB1 : offB0,
                     (unsigned)K, (unsigned)N, TK, TN, (unsigned)K);
  };

  auto compute = [&](int buf) {
    v16bf aF[2], bF[4];
#pragma unroll
    for (int ms = 0; ms < 2; ++ms) {
      const int base = (mOff + ms * 16 + mr) * LDA + g * 8;
      bf16x8 lo = *(const bf16x8*)&ldsA[buf][base];
      bf16x8 hi = *(const bf16x8*)&ldsA[buf][base + 16];
      aF[ms] = __builtin_shufflevector(lo, hi,
                 0,1,2,3,4,5,6,7,8,9,10,11,12,13,14,15);
    }
#pragma unroll
    for (int ns = 0; ns < 4; ++ns) {
      const int base = (nOff + ns * 16 + mr) * LDA + g * 8;
      bf16x8 lo = *(const bf16x8*)&ldsB[buf][base];
      bf16x8 hi = *(const bf16x8*)&ldsB[buf][base + 16];
      bF[ns] = __builtin_shufflevector(lo, hi,
                 0,1,2,3,4,5,6,7,8,9,10,11,12,13,14,15);
    }
#pragma unroll
    for (int ms = 0; ms < 2; ++ms)
#pragma unroll
      for (int ns = 0; ns < 4; ++ns)
        acc[ms][ns] = __builtin_amdgcn_wmma_f32_16x16x32_bf16(
            false, aF[ms], false, bF[ns], (short)0, acc[ms][ns], false, false);
  };

  const int nIter = K / TK;

  if (wave == 0) issue_tiles(0, 0);           // prologue: tile 0 -> buf 0

  for (int i = 0; i < nIter; ++i) {
    const int cur = i & 1;
    if (wave == 0) {
      if (i + 1 < nIter) {
        issue_tiles(1 - cur, (i + 1) * TK);   // buf^1 free since barrier of i-1
        __builtin_amdgcn_s_wait_tensorcnt(2); // tile i landed (in-order retire)
      } else {
        __builtin_amdgcn_s_wait_tensorcnt(0);
      }
    }
    __syncthreads();                          // release tile i to all waves
    compute(cur);                             // 12x ds_load_b128 + 8x v_wmma
    __syncthreads();                          // buf cur reusable next iteration
  }

#pragma unroll
  for (int ms = 0; ms < 2; ++ms) {
    const int mBase = mTile + mOff + ms * 16 + g * 8;
#pragma unroll
    for (int ns = 0; ns < 4; ++ns) {
      const int nc = nTile + nOff + ns * 16 + mr;
      const float bb = bias[nc];
#pragma unroll
      for (int r = 0; r < 8; ++r)
        y[(size_t)(mBase + r) * N + nc] = acc[ms][ns][r] + bb;
    }
  }
}

// =============================================================================
// Fallback: fused dequant + GEMM (round-2 kernel) if workspace is too small.
// =============================================================================
__global__ __launch_bounds__(256)
void w4a16_gemm_fused(const float* __restrict__ x,
                      const int*   __restrict__ w,
                      const float* __restrict__ scale,
                      const float* __restrict__ offs,
                      const float* __restrict__ bias,
                      float*       __restrict__ y,
                      int M, int N, int K)
{
  __shared__ __align__(16) unsigned short ldsA[2][TM * LDA];
  __shared__ __align__(16) unsigned short ldsB[2][TN * LDA];

  const int tid  = threadIdx.x;
  const int wave = tid >> 5;
  const int lane = tid & 31;

  const int mTile = blockIdx.y * TM;
  const int nTile = blockIdx.x * TN;

  const int mOff = (wave & 3) * 32;
  const int nOff = (wave >> 2) * 64;

  const int ldRow = tid >> 1;
  const int ldCol = (tid & 1) * 16;

  const float* xRow = x + (size_t)(mTile + ldRow) * K + ldCol;
  const int*   wRow = w + (size_t)(nTile + ldRow) * K + ldCol;
  const float  sc   = scale[nTile + ldRow];
  const float  of   = offs [nTile + ldRow];

  const int g  = lane >> 4;
  const int mr = lane & 15;

  v8f acc[2][4] = {};
  f4 xr[4];
  i4 qr[4];

  auto gload = [&](int k0) {
#pragma unroll
    for (int i = 0; i < 4; ++i) {
      xr[i] = *(const f4*)(xRow + k0 + i * 4);
      qr[i] = *(const i4*)(wRow + k0 + i * 4);
    }
  };
  auto commit = [&](int buf) {
    us8 av[2], bv[2];
#pragma unroll
    for (int i = 0; i < 16; ++i)
      av[i >> 3][i & 7] = f2bf(xr[i >> 2][i & 3]);
#pragma unroll
    for (int i = 0; i < 16; ++i)
      bv[i >> 3][i & 7] = f2bf(((float)qr[i >> 2][i & 3] + of) * sc);
    const int base = ldRow * LDA + ldCol;
    *(us8*)&ldsA[buf][base]     = av[0];
    *(us8*)&ldsA[buf][base + 8] = av[1];
    *(us8*)&ldsB[buf][base]     = bv[0];
    *(us8*)&ldsB[buf][base + 8] = bv[1];
  };
  auto compute = [&](int buf) {
    v16bf aF[2], bF[4];
#pragma unroll
    for (int ms = 0; ms < 2; ++ms) {
      const int base = (mOff + ms * 16 + mr) * LDA + g * 8;
      bf16x8 lo = *(const bf16x8*)&ldsA[buf][base];
      bf16x8 hi = *(const bf16x8*)&ldsA[buf][base + 16];
      aF[ms] = __builtin_shufflevector(lo, hi,
                 0,1,2,3,4,5,6,7,8,9,10,11,12,13,14,15);
    }
#pragma unroll
    for (int ns = 0; ns < 4; ++ns) {
      const int base = (nOff + ns * 16 + mr) * LDA + g * 8;
      bf16x8 lo = *(const bf16x8*)&ldsB[buf][base];
      bf16x8 hi = *(const bf16x8*)&ldsB[buf][base + 16];
      bF[ns] = __builtin_shufflevector(lo, hi,
                 0,1,2,3,4,5,6,7,8,9,10,11,12,13,14,15);
    }
#pragma unroll
    for (int ms = 0; ms < 2; ++ms)
#pragma unroll
      for (int ns = 0; ns < 4; ++ns)
        acc[ms][ns] = __builtin_amdgcn_wmma_f32_16x16x32_bf16(
            false, aF[ms], false, bF[ns], (short)0, acc[ms][ns], false, false);
  };

  const int nIter = K / TK;
  gload(0);
  commit(0);
  __syncthreads();

  for (int i = 0; i < nIter; ++i) {
    const int cur = i & 1;
    if (i + 1 < nIter) {
      gload((i + 1) * TK);
      if (i + 2 < nIter) {
        __builtin_prefetch(xRow + (size_t)(i + 2) * TK, 0, 1);
        __builtin_prefetch(wRow + (size_t)(i + 2) * TK, 0, 1);
      }
    }
    compute(cur);
    if (i + 1 < nIter)
      commit(1 - cur);
    __syncthreads();
  }

#pragma unroll
  for (int ms = 0; ms < 2; ++ms) {
    const int mBase = mTile + mOff + ms * 16 + g * 8;
#pragma unroll
    for (int ns = 0; ns < 4; ++ns) {
      const int nc = nTile + nOff + ns * 16 + mr;
      const float bb = bias[nc];
#pragma unroll
      for (int r = 0; r < 8; ++r)
        y[(size_t)(mBase + r) * N + nc] = acc[ms][ns][r] + bb;
    }
  }
}

// =============================================================================
extern "C" void kernel_launch(void* const* d_in, const int* in_sizes, int n_in,
                              void* d_out, int out_size, void* d_ws, size_t ws_size,
                              hipStream_t stream) {
  const float* x  = (const float*)d_in[0];   // [B,S,K] fp32
  const int*   w  = (const int*)  d_in[1];   // [N,K] int4-in-int32
  const float* sc = (const float*)d_in[2];   // [N,1]
  const float* of = (const float*)d_in[3];   // [N,1] (pre-negated)
  const float* bs = (const float*)d_in[4];   // [N]
  float* y = (float*)d_out;                  // [B,S,N] fp32

  const int N = in_sizes[4];                 // 11008
  const int K = in_sizes[1] / N;             // 4096
  const int M = in_sizes[0] / K;             // B*S = 4096

  const size_t xbBytes = (size_t)M * K * sizeof(unsigned short);
  const size_t wbBytes = (size_t)N * K * sizeof(unsigned short);

  dim3 grid(N / TN, M / TM);                 // 86 x 32 blocks
  dim3 block(256);

  if (ws_size >= xbBytes + wbBytes) {
    unsigned short* xb = (unsigned short*)d_ws;
    unsigned short* wb = (unsigned short*)((char*)d_ws + xbBytes);

    const long long xTotal = (long long)M * K;
    dim3 gX((unsigned)((xTotal / 8 + 255) / 256));
    hipLaunchKernelGGL(cvt_x_bf16, gX, block, 0, stream, x, xb, xTotal);

    dim3 gW((unsigned)(K / (256 * 8)) ? (unsigned)(K / (256 * 8)) : 1u,
            (unsigned)N);
    hipLaunchKernelGGL(dequant_w_bf16, gW, block, 0, stream, w, sc, of, wb, K);

    hipLaunchKernelGGL(gemm_bf16_tdm, grid, block, 0, stream,
                       xb, wb, bs, y, M, N, K);
  } else {
    hipLaunchKernelGGL(w4a16_gemm_fused, grid, block, 0, stream,
                       x, w, sc, of, bs, y, M, N, K);
  }
}
